// FLAME_81149112090920
// MI455X (gfx1250) — matrix operand order repaired and checked by reading, not compiled, not run on hardware
//
#include <hip/hip_runtime.h>
#include <hip/hip_bf16.h>

// ---------------- problem constants ----------------
#define B_    512
#define V_    5023
#define NF_   9976
#define NJ_   5
#define NB_   400
#define NP_   36
#define NV3_  (V_*3)          // 15069
#define KC_   (NB_ + NP_)     // 436
#define KPAD_ 448             // K padded to multiple of 32
#define KT_   (KPAD_/32)      // 14 k-tiles
#define NPAD_ 15104           // N padded to multiple of 64 (so every wave owns 4 full tiles)
#define NT_   (NPAD_/16)      // 944 n-tiles, multiple of 4
#define MT_   (B_/16)         // 32 m-tiles

typedef __attribute__((ext_vector_type(16))) _Float16 v16h;
typedef __attribute__((ext_vector_type(8)))  float    v8f;

// ========== 1. Rodrigues: pose (B,NJ,3) -> rot_mats (B,NJ,9) ==========
__global__ __launch_bounds__(256)
void k_rodrigues(const float* __restrict__ pose, float* __restrict__ rot) {
    int idx = blockIdx.x * blockDim.x + threadIdx.x;
    if (idx >= B_ * NJ_) return;
    float x = pose[idx*3+0], y = pose[idx*3+1], z = pose[idx*3+2];
    float xe = x + 1e-8f, ye = y + 1e-8f, ze = z + 1e-8f;
    float angle = sqrtf(xe*xe + ye*ye + ze*ze);
    float inv = 1.0f / angle;
    float rx = x*inv, ry = y*inv, rz = z*inv;
    float c = cosf(angle), s = sinf(angle), t = 1.0f - c;
    float* R = rot + idx*9;
    R[0] = 1.0f - t*(ry*ry+rz*rz);
    R[1] = -s*rz + t*rx*ry;
    R[2] =  s*ry + t*rx*rz;
    R[3] =  s*rz + t*rx*ry;
    R[4] = 1.0f - t*(rx*rx+rz*rz);
    R[5] = -s*rx + t*ry*rz;
    R[6] = -s*ry + t*rx*rz;
    R[7] =  s*rx + t*ry*rz;
    R[8] = 1.0f - t*(rx*rx+ry*ry);
}

// ========== 2. Build A fragments (f16, WMMA 16x32 layout) ==========
// A row m = batch b, col k = [betas(400) | pose_feature(36) | pad]
// lane L: m = L&15 ; VGPR half i: k = (i<8) ? (L>>4)*8+i : 16+(L>>4)*8+(i-8)
__global__ __launch_bounds__(256)
void k_build_A(const float* __restrict__ betas, const float* __restrict__ rot,
               _Float16* __restrict__ At) {
    int idx = blockIdx.x * blockDim.x + threadIdx.x;
    if (idx >= KT_ * MT_ * 512) return;
    int kt   = idx / (MT_ * 512);
    int mt   = (idx / 512) % MT_;
    int e    = idx & 511;
    int lane = e >> 4;
    int i    = e & 15;
    int m     = mt*16 + (lane & 15);
    int khalf = (lane >> 4) * 8;
    int kin   = (i < 8) ? (khalf + i) : (16 + khalf + (i - 8));
    int k     = kt*32 + kin;
    float v = 0.0f;
    if (k < NB_) {
        v = betas[m*NB_ + k];
    } else if (k < KC_) {
        int p = k - NB_;
        int j = p / 9 + 1;
        int t = p % 9;
        float d = (t == 0 || t == 4 || t == 8) ? 1.0f : 0.0f;
        v = rot[(m*NJ_ + j)*9 + t] - d;
    }
    At[idx] = (_Float16)v;
}

// ========== 3. Build B fragments (f16, WMMA 32x16 layout) ==========
// B[k][n]: k<400 -> shapedirs[n*400+k]; 400<=k<436 -> posedirs[(k-400)*15069+n]
// lane L: n = L&15 ; half i: k = (i<8) ? (L>>4)*8+i : 16+(L>>4)*8+(i-8)
__global__ __launch_bounds__(256)
void k_build_B(const float* __restrict__ shapedirs, const float* __restrict__ posedirs,
               _Float16* __restrict__ Bt) {
    int idx = blockIdx.x * blockDim.x + threadIdx.x;
    if (idx >= KT_ * NT_ * 512) return;
    int kt   = idx / (NT_ * 512);
    int nt   = (idx / 512) % NT_;
    int e    = idx & 511;
    int lane = e >> 4;
    int i    = e & 15;
    int n = nt*16 + (lane & 15);
    int k = kt*32 + ((i < 8) ? ((lane >> 4)*8 + i) : (16 + (lane >> 4)*8 + (i - 8)));
    float v = 0.0f;
    if (n < NV3_) {
        if (k < NB_)       v = shapedirs[n*NB_ + k];
        else if (k < KC_)  v = posedirs[(k - NB_)*NV3_ + n];
    }
    Bt[idx] = (_Float16)v;
}

// ========== 4. S = JR . shapedirs  (15 x 400) and Jt = JR . v_template ==========
__global__ __launch_bounds__(256)
void k_reduceS(const float* __restrict__ JR, const float* __restrict__ shapedirs,
               const float* __restrict__ vt, float* __restrict__ S, float* __restrict__ Jt) {
    int t = blockIdx.x * blockDim.x + threadIdx.x;
    if (t < NJ_*3*NB_) {
        int j = t / (3*NB_);
        int c = (t / NB_) % 3;
        int l = t % NB_;
        const float* jr = JR + j*V_;
        float acc = 0.0f;
        for (int v = 0; v < V_; ++v)
            acc += jr[v] * shapedirs[v*(3*NB_) + c*NB_ + l];
        S[t] = acc;
    } else if (t < NJ_*3*NB_ + NJ_*3) {
        int u = t - NJ_*3*NB_;
        int j = u / 3, c = u % 3;
        const float* jr = JR + j*V_;
        float acc = 0.0f;
        for (int v = 0; v < V_; ++v)
            acc += jr[v] * vt[v*3 + c];
        Jt[u] = acc;
    }
}

// ========== 5. Joints + rigid-transform chain (1 thread / batch) ==========
__global__ __launch_bounds__(256)
void k_joints(const float* __restrict__ betas, const float* __restrict__ S,
              const float* __restrict__ Jt, const float* __restrict__ rot,
              float* __restrict__ outJ, float* __restrict__ Arel) {
    int b = blockIdx.x * blockDim.x + threadIdx.x;
    if (b >= B_) return;
    const float* be = betas + b*NB_;
    float J[NJ_][3];
    for (int j = 0; j < NJ_; ++j)
        for (int c = 0; c < 3; ++c) {
            const float* s = S + (j*3 + c)*NB_;
            float acc = Jt[j*3 + c];
            for (int l = 0; l < NB_; ++l) acc += be[l] * s[l];
            J[j][c] = acc;
        }
    const int par[NJ_] = {-1, 0, 1, 1, 1};
    float chain[NJ_][16];
    for (int j = 0; j < NJ_; ++j) {
        float rel[3];
        if (j == 0) { rel[0]=J[0][0]; rel[1]=J[0][1]; rel[2]=J[0][2]; }
        else for (int c = 0; c < 3; ++c) rel[c] = J[j][c] - J[par[j]][c];
        const float* R = rot + (b*NJ_ + j)*9;
        float M[16] = { R[0],R[1],R[2],rel[0],
                        R[3],R[4],R[5],rel[1],
                        R[6],R[7],R[8],rel[2],
                        0.f, 0.f, 0.f, 1.f };
        if (j == 0) {
            for (int q = 0; q < 16; ++q) chain[0][q] = M[q];
        } else {
            const float* P = chain[par[j]];
            for (int r = 0; r < 4; ++r)
                for (int c2 = 0; c2 < 4; ++c2) {
                    float acc = 0.f;
                    for (int q = 0; q < 4; ++q) acc += P[r*4+q] * M[q*4+c2];
                    chain[j][r*4+c2] = acc;
                }
        }
        outJ[b*NJ_*3 + j*3 + 0] = chain[j][3];
        outJ[b*NJ_*3 + j*3 + 1] = chain[j][7];
        outJ[b*NJ_*3 + j*3 + 2] = chain[j][11];
        float* A = Arel + b*(NJ_*16) + j*16;
        for (int q = 0; q < 16; ++q) A[q] = chain[j][q];
        for (int r = 0; r < 3; ++r)
            A[r*4+3] = chain[j][r*4+3]
                     - (chain[j][r*4+0]*J[j][0] + chain[j][r*4+1]*J[j][1] + chain[j][r*4+2]*J[j][2]);
    }
}

// ========== 6. Main GEMM: v_posed = v_template + A(512xK) * B(KxN), WMMA f16 ==========
// 8 waves/block; each wave owns 4 consecutive 16x16 N-tiles (A reuse x4).
// NT_ is a multiple of 4, so no per-tile guards: clean back-to-back WMMAs.
__global__ __launch_bounds__(256)
void k_gemm(const _Float16* __restrict__ At, const _Float16* __restrict__ Bt,
            const float* __restrict__ vt, float* __restrict__ vposed) {
    const int lane  = threadIdx.x & 31;
    // wave-uniform tile coordinates, forced into SGPRs
    const int nt0   = __builtin_amdgcn_readfirstlane(
                        (blockIdx.x * 8 + (threadIdx.x >> 5)) * 4);
    const int mtile = blockIdx.y;
    if (nt0 >= NT_) return;   // scalar branch (nt0 is SGPR)

    v8f acc0 = {}, acc1 = {}, acc2 = {}, acc3 = {};
    for (int kt = 0; kt < KT_; ++kt) {
        v16h a  = *(const v16h*)(At + (kt*MT_ + mtile)*512 + lane*16);
        const _Float16* bp = Bt + (kt*NT_ + nt0)*512 + lane*16;
        v16h b0 = *(const v16h*)(bp);
        v16h b1 = *(const v16h*)(bp + 512);
        v16h b2 = *(const v16h*)(bp + 1024);
        v16h b3 = *(const v16h*)(bp + 1536);
        acc0 = __builtin_amdgcn_wmma_f32_16x16x32_f16(false, a, false, b0,
                                                      (short)0, acc0, false, false);
        acc1 = __builtin_amdgcn_wmma_f32_16x16x32_f16(false, a, false, b1,
                                                      (short)0, acc1, false, false);
        acc2 = __builtin_amdgcn_wmma_f32_16x16x32_f16(false, a, false, b2,
                                                      (short)0, acc2, false, false);
        acc3 = __builtin_amdgcn_wmma_f32_16x16x32_f16(false, a, false, b3,
                                                      (short)0, acc3, false, false);
    }
    // C/D layout: lane L -> n = L&15, rows m = r + (L>>4)*8
    const int mbase = mtile*16 + (lane >> 4)*8;
    const int nlow  = lane & 15;
    {
        int ncol = (nt0 + 0)*16 + nlow;
        if (ncol < NV3_) {
            float tv = vt[ncol];
            for (int r = 0; r < 8; ++r) vposed[(mbase + r)*NV3_ + ncol] = tv + acc0[r];
        }
    }
    {
        int ncol = (nt0 + 1)*16 + nlow;
        if (ncol < NV3_) {
            float tv = vt[ncol];
            for (int r = 0; r < 8; ++r) vposed[(mbase + r)*NV3_ + ncol] = tv + acc1[r];
        }
    }
    {
        int ncol = (nt0 + 2)*16 + nlow;
        if (ncol < NV3_) {
            float tv = vt[ncol];
            for (int r = 0; r < 8; ++r) vposed[(mbase + r)*NV3_ + ncol] = tv + acc2[r];
        }
    }
    {
        int ncol = (nt0 + 3)*16 + nlow;
        if (ncol < NV3_) {
            float tv = vt[ncol];
            for (int r = 0; r < 8; ++r) vposed[(mbase + r)*NV3_ + ncol] = tv + acc3[r];
        }
    }
}

// ========== 7. zero normal accumulator ==========
__global__ __launch_bounds__(256)
void k_zero(float4* __restrict__ p, int n4) {
    int idx = blockIdx.x * blockDim.x + threadIdx.x;
    if (idx < n4) p[idx] = make_float4(0.f, 0.f, 0.f, 0.f);
}

// ========== 8. Face normal scatter (atomics) ==========
__global__ __launch_bounds__(256)
void k_faces(const int* __restrict__ faces, const float* __restrict__ vp,
             float* __restrict__ nacc) {
    int idx = blockIdx.x * blockDim.x + threadIdx.x;
    if (idx >= B_ * NF_) return;
    int f = idx % NF_, b = idx / NF_;
    int i0 = faces[f*3+0], i1 = faces[f*3+1], i2 = faces[f*3+2];
    const float* base = vp + (size_t)b * NV3_;
    float a0 = base[i0*3], a1 = base[i0*3+1], a2 = base[i0*3+2];
    float b0 = base[i1*3], b1 = base[i1*3+1], b2 = base[i1*3+2];
    float c0 = base[i2*3], c1 = base[i2*3+1], c2 = base[i2*3+2];
    float* nb = nacc + (size_t)b * NV3_;
    // at i1 : cross(v2-v1, v0-v1)
    {
        float ux = c0-b0, uy = c1-b1, uz = c2-b2;
        float vx = a0-b0, vy = a1-b1, vz = a2-b2;
        atomicAdd(&nb[i1*3+0], uy*vz - uz*vy);
        atomicAdd(&nb[i1*3+1], uz*vx - ux*vz);
        atomicAdd(&nb[i1*3+2], ux*vy - uy*vx);
    }
    // at i2 : cross(v0-v2, v1-v2)
    {
        float ux = a0-c0, uy = a1-c1, uz = a2-c2;
        float vx = b0-c0, vy = b1-c1, vz = b2-c2;
        atomicAdd(&nb[i2*3+0], uy*vz - uz*vy);
        atomicAdd(&nb[i2*3+1], uz*vx - ux*vz);
        atomicAdd(&nb[i2*3+2], ux*vy - uy*vx);
    }
    // at i0 : cross(v1-v0, v2-v0)
    {
        float ux = b0-a0, uy = b1-a1, uz = b2-a2;
        float vx = c0-a0, vy = c1-a1, vz = c2-a2;
        atomicAdd(&nb[i0*3+0], uy*vz - uz*vy);
        atomicAdd(&nb[i0*3+1], uz*vx - ux*vz);
        atomicAdd(&nb[i0*3+2], ux*vy - uy*vx);
    }
}

// ========== 9. normalize + normal coord system -> d_out (ncs section) ==========
__global__ __launch_bounds__(256)
void k_ncs(const float* __restrict__ nacc, float* __restrict__ outN) {
    int idx = blockIdx.x * blockDim.x + threadIdx.x;
    if (idx >= B_ * V_) return;
    float nx = nacc[idx*3+0], ny = nacc[idx*3+1], nz = nacc[idx*3+2];
    float len = sqrtf(nx*nx + ny*ny + nz*nz);
    float inv = 1.0f / fmaxf(len, 1e-6f);
    float n0 = nx*inv, n1 = ny*inv, n2 = nz*inv;
    float l2 = sqrtf(n0*n0 + n1*n1 + n2*n2);
    float inv2 = 1.0f / fmaxf(l2, 1e-12f);
    float ez0 = n0*inv2, ez1 = n1*inv2, ez2 = n2*inv2;
    // ex = normalize(cross([0,1,0], ez)) = normalize([ez.z, 0, -ez.x])
    float ex0 = ez2, ex2 = -ez0;
    float le = sqrtf(ex0*ex0 + ex2*ex2);
    float inve = 1.0f / fmaxf(le, 1e-12f);
    ex0 *= inve; ex2 *= inve;
    // ey = cross(ez, ex)  (ex1 == 0)
    float ey0 =  ez1*ex2;
    float ey1 =  ez2*ex0 - ez0*ex2;
    float ey2 = -ez1*ex0;
    float* o = outN + (size_t)idx * 9;
    o[0]=ex0; o[1]=0.f; o[2]=ex2;
    o[3]=ey0; o[4]=ey1; o[5]=ey2;
    o[6]=ez0; o[7]=ez1; o[8]=ez2;
}

// ========== 10. Skinning -> d_out (verts section) ==========
__global__ __launch_bounds__(256)
void k_skin(const float* __restrict__ w, const float* __restrict__ Arel,
            const float* __restrict__ vp, float* __restrict__ verts) {
    int idx = blockIdx.x * blockDim.x + threadIdx.x;
    if (idx >= B_ * V_) return;
    int v = idx % V_, b = idx / V_;
    const float* W = w + v*NJ_;
    const float* A = Arel + b*(NJ_*16);
    float T[12];
    for (int q = 0; q < 12; ++q) {
        float acc = 0.f;
        for (int j = 0; j < NJ_; ++j) acc += W[j] * A[j*16 + q];
        T[q] = acc;
    }
    float x = vp[(size_t)b*NV3_ + v*3 + 0];
    float y = vp[(size_t)b*NV3_ + v*3 + 1];
    float z = vp[(size_t)b*NV3_ + v*3 + 2];
    verts[idx*3+0] = T[0]*x + T[1]*y + T[2]*z  + T[3];
    verts[idx*3+1] = T[4]*x + T[5]*y + T[6]*z  + T[7];
    verts[idx*3+2] = T[8]*x + T[9]*y + T[10]*z + T[11];
}

// =====================================================================
static inline char* ws_take(char*& cur, size_t bytes) {
    char* p = cur;
    size_t aligned = (bytes + 255) & ~(size_t)255;
    cur += aligned;
    return p;
}

extern "C" void kernel_launch(void* const* d_in, const int* in_sizes, int n_in,
                              void* d_out, int out_size, void* d_ws, size_t ws_size,
                              hipStream_t stream) {
    (void)in_sizes; (void)n_in; (void)out_size; (void)ws_size;
    const float* betas     = (const float*)d_in[0];
    const float* pose      = (const float*)d_in[1];
    const float* v_tmpl    = (const float*)d_in[2];
    const float* shapedirs = (const float*)d_in[3];
    const float* posedirs  = (const float*)d_in[4];
    const float* JR        = (const float*)d_in[5];
    const float* lbs_w     = (const float*)d_in[6];
    const int*   faces     = (const int*)d_in[7];

    float* out = (float*)d_out;
    float* out_verts = out;                              // (B,V,3)
    float* out_J     = out + (size_t)B_*V_*3;            // (B,NJ,3)
    float* out_ncs   = out_J + (size_t)B_*NJ_*3;         // (B,V,3,3)

    char* cur = (char*)d_ws;
    float*    ws_vposed = (float*)   ws_take(cur, (size_t)B_*NV3_*4);      // 30.9 MB
    float*    ws_nacc   = (float*)   ws_take(cur, (size_t)B_*NV3_*4);      // 30.9 MB
    _Float16* ws_At     = (_Float16*)ws_take(cur, (size_t)KT_*MT_*512*2);  // 0.46 MB
    _Float16* ws_Bt     = (_Float16*)ws_take(cur, (size_t)KT_*NT_*512*2);  // 13.5 MB
    float*    ws_rot    = (float*)   ws_take(cur, (size_t)B_*NJ_*9*4);
    float*    ws_S      = (float*)   ws_take(cur, (size_t)NJ_*3*NB_*4);
    float*    ws_Jt     = (float*)   ws_take(cur, (size_t)NJ_*3*4);
    float*    ws_A      = (float*)   ws_take(cur, (size_t)B_*NJ_*16*4);

    const int TB = 256;
    #define NBLK(n) (((n) + TB - 1) / TB)

    k_rodrigues<<<NBLK(B_*NJ_), TB, 0, stream>>>(pose, ws_rot);
    k_build_A  <<<NBLK(KT_*MT_*512), TB, 0, stream>>>(betas, ws_rot, ws_At);
    k_build_B  <<<NBLK(KT_*NT_*512), TB, 0, stream>>>(shapedirs, posedirs, ws_Bt);
    k_reduceS  <<<NBLK(NJ_*3*NB_ + NJ_*3), TB, 0, stream>>>(JR, shapedirs, v_tmpl, ws_S, ws_Jt);
    k_joints   <<<NBLK(B_), TB, 0, stream>>>(betas, ws_S, ws_Jt, ws_rot, out_J, ws_A);

    dim3 ggrid((NT_ + 31) / 32, MT_);   // 8 waves/block * 4 n-tiles/wave = 32 n-tiles/block
    k_gemm<<<ggrid, TB, 0, stream>>>(ws_At, ws_Bt, v_tmpl, ws_vposed);

    int n4 = (B_*NV3_) / 4;
    k_zero <<<NBLK(n4), TB, 0, stream>>>((float4*)ws_nacc, n4);
    k_faces<<<NBLK(B_*NF_), TB, 0, stream>>>(faces, ws_vposed, ws_nacc);
    k_ncs  <<<NBLK(B_*V_), TB, 0, stream>>>(ws_nacc, out_ncs);
    k_skin <<<NBLK(B_*V_), TB, 0, stream>>>(lbs_w, ws_A, ws_vposed, out_verts);
    #undef NBLK
}
